// ColumnAttention_42331197669728
// MI455X (gfx1250) — matrix operand order repaired and checked by reading, not compiled, not run on hardware
//
#include <hip/hip_runtime.h>
#include <hip/hip_bf16.h>

// Problem dims (fixed by the reference)
#define N_BATCH 32
#define C_DIM   64
#define H_DIM   128
#define W_DIM   128
#define QK_DIM  32
#define SPITCH  132          // padded f32 score-row stride (16B-aligned rows)
#define NTHREADS 128         // 4 wave32 per workgroup

typedef __attribute__((ext_vector_type(16))) __bf16 v16bf;
typedef __attribute__((ext_vector_type(8)))  __bf16 v8bf;
typedef __attribute__((ext_vector_type(8)))  float  v8f;

static __device__ __forceinline__ v8f wmma_bf16(v16bf a, v16bf b, v8f c) {
  // D(f32 16x16) = A(bf16 16x32) * B(bf16 32x16) + C
  return __builtin_amdgcn_wmma_f32_16x16x32_bf16(
      /*neg_a=*/false, a, /*neg_b=*/false, b,
      /*c_mod=*/(short)0, c, /*reuse_a=*/false, /*reuse_b=*/false);
}

// ---- CDNA5 WMMA fragment loaders (wave32 layouts per ISA 7.12.2) ----------
// All layouts below are arranged so K is contiguous: each fragment is two
// 16-byte LDS reads (ds_load_b128), no scalar u16 gathering.

// A 16x32 bf16, row-major with row stride rs (elements), K contiguous.
// lane half 0 holds K {0..7, 16..23}, half 1 holds K {8..15, 24..31}.
static __device__ __forceinline__ v16bf
load_a_rm(const __bf16* p, int row0, int col0, int rs, int lane) {
  const int m = lane & 15, half = lane >> 4;
  const __bf16* base = p + (row0 + m) * rs + col0 + half * 8;
  v8bf lo = *(const v8bf*)(base);        // K = half*8 + 0..7
  v8bf hi = *(const v8bf*)(base + 16);   // K = 16 + half*8 + 0..7
  return __builtin_shufflevector(lo, hi, 0, 1, 2, 3, 4, 5, 6, 7,
                                 8, 9, 10, 11, 12, 13, 14, 15);
}

// A 16x32 from f32 source (scores), K contiguous; converts to bf16.
static __device__ __forceinline__ v16bf
load_a_f32_rm(const float* p, int row0, int col0, int rs, int lane) {
  const int m = lane & 15, half = lane >> 4;
  const float* base = p + (row0 + m) * rs + col0 + half * 8;
  v16bf a;
#pragma unroll
  for (int e = 0; e < 8; ++e) {
    a[e]     = (__bf16)base[e];        // K = half*8 + e
    a[e + 8] = (__bf16)base[16 + e];   // K = 16 + half*8 + e
  }
  return a;
}

// B 32x16 bf16 from column-major storage: element (k,n) at p[(col0+n)*ld + row0 + k].
// lane holds column n = lane&15; half 0 -> K {0..7, 16..23}, half 1 -> K {8..15, 24..31}.
static __device__ __forceinline__ v16bf
load_b_ct(const __bf16* p, int row0, int col0, int ld, int lane) {
  const int n = lane & 15, half = lane >> 4;
  const __bf16* base = p + (col0 + n) * ld + row0 + half * 8;
  v8bf lo = *(const v8bf*)(base);        // K = half*8 + 0..7
  v8bf hi = *(const v8bf*)(base + 16);   // K = 16 + half*8 + 0..7
  return __builtin_shufflevector(lo, hi, 0, 1, 2, 3, 4, 5, 6, 7,
                                 8, 9, 10, 11, 12, 13, 14, 15);
}

// ---------------------------------------------------------------------------
// Dynamic LDS map (98 KB total):
//   region0 [0, 67584):  phase 0/1: sXT[H][C] bf16 (16K), sWq (4K), sWk (4K),
//                        sWv (8K);  phase >=2: sS[H][SPITCH] f32 (aliased).
//   sV  [C][H]  bf16 16K     sQT [H][QK] bf16 8K     sKT [H][QK] bf16 8K
#define REGION0_BYTES ((size_t)H_DIM * SPITCH * sizeof(float))   // 67584
#define SMEM_BYTES (REGION0_BYTES + (size_t)(C_DIM * H_DIM + 2 * H_DIM * QK_DIM) * 2)

__global__ void column_attention_kernel(const float* __restrict__ x,
                                        const float* __restrict__ wq,
                                        const float* __restrict__ wk,
                                        const float* __restrict__ wv,
                                        const float* __restrict__ gama,
                                        float* __restrict__ out) {
  extern __shared__ char smem[];
  // region0, phase 0/1 view:
  __bf16* sXT = (__bf16*)smem;                    // [H][C]  (h major, c contig)
  __bf16* sWq = sXT + H_DIM * C_DIM;              // [QK][C]
  __bf16* sWk = sWq + QK_DIM * C_DIM;             // [QK][C]
  __bf16* sWv = sWk + QK_DIM * C_DIM;             // [C][C]
  // region0, phase >=2 view (aliased; written only after barrier):
  float*  sS  = (float*)smem;                     // [H][SPITCH]
  // persistent buffers:
  __bf16* sV  = (__bf16*)(smem + REGION0_BYTES);  // [C][H]  (c major, h contig)
  __bf16* sQT = sV  + C_DIM * H_DIM;              // [H][QK] (i major, q contig)
  __bf16* sKT = sQT + H_DIM * QK_DIM;             // [H][QK] (j major, q contig)

  const int tid  = threadIdx.x;
  const int lane = tid & 31;
  const int wave = tid >> 5;
  const int hl   = lane >> 4;          // half-wave select
  const int ln   = lane & 15;
  const int n    = blockIdx.x >> 7;    // batch
  const int w    = blockIdx.x & 127;   // image column
  const float g  = gama[0];

  // ---- Phase 0: stage x-column (transposed) + weights into LDS as bf16 ----
  {
    const float* xn = x + (size_t)n * C_DIM * H_DIM * W_DIM + w;
#pragma unroll 4
    for (int i = tid; i < C_DIM * H_DIM; i += NTHREADS) {
      const int c = i >> 7, h = i & 127;            // i = c*H + h
      sXT[h * C_DIM + c] = (__bf16)xn[(size_t)i * W_DIM];  // L2-resident gather
    }
    for (int i = tid; i < QK_DIM * C_DIM; i += NTHREADS) {
      sWq[i] = (__bf16)wq[i];
      sWk[i] = (__bf16)wk[i];
    }
    for (int i = tid; i < C_DIM * C_DIM; i += NTHREADS)
      sWv[i] = (__bf16)wv[i];
  }
  __syncthreads();

  // ---- Phase 1a: Q = wq*X, K = wk*X (2x8 tiles each); store transposed ----
  for (int t = wave * 8; t < wave * 8 + 8; ++t) {   // waves 0,1 -> Q; 2,3 -> K
    const int isK = t >> 4;
    const int mt = (t & 15) >> 3, nt = t & 7;
    const __bf16* Wm = isK ? sWk : sWq;
    v8f acc = {};
#pragma unroll
    for (int kk = 0; kk < C_DIM; kk += 32) {
      v16bf a = load_a_rm(Wm, mt * 16, kk, C_DIM, lane);
      v16bf b = load_b_ct(sXT, kk, nt * 16, C_DIM, lane);  // B[c][h]=sXT[h*C+c]
      acc = wmma_bf16(a, b, acc);
    }
    // D rows r -> 8 consecutive q; pack & store 16B per lane (ds_store_b128).
    __bf16* dstT = isK ? sKT : sQT;
    v8bf pk;
#pragma unroll
    for (int r = 0; r < 8; ++r) pk[r] = (__bf16)acc[r];
    *(v8bf*)&dstT[(nt * 16 + ln) * QK_DIM + mt * 16 + hl * 8] = pk;
  }

  // ---- Phase 1b: V = wv*X (4x8 tiles); store [c][h] (h contig) ------------
  for (int t = wave * 8; t < wave * 8 + 8; ++t) {
    const int mt = t >> 3, nt = t & 7;
    v8f acc = {};
#pragma unroll
    for (int kk = 0; kk < C_DIM; kk += 32) {
      v16bf a = load_a_rm(sWv, mt * 16, kk, C_DIM, lane);
      v16bf b = load_b_ct(sXT, kk, nt * 16, C_DIM, lane);
      acc = wmma_bf16(a, b, acc);
    }
#pragma unroll
    for (int r = 0; r < 8; ++r)
      sV[(mt * 16 + hl * 8 + r) * H_DIM + nt * 16 + ln] = (__bf16)acc[r];
  }
  __syncthreads();   // region0 reads done; safe to overwrite with scores

  // ---- Phase 2: S = Q^T * K  (8x8 tiles, one K-step of 32) ----------------
  for (int t = wave * 16; t < wave * 16 + 16; ++t) {
    const int it = t >> 3, jt = t & 7;
    v16bf a = load_a_rm(sQT, it * 16, 0, QK_DIM, lane);   // A[i][q], q contig
    v16bf b = load_b_ct(sKT, 0, jt * 16, QK_DIM, lane);   // B[q][j], q contig
    v8f acc = {};
    acc = wmma_bf16(a, b, acc);
#pragma unroll
    for (int r = 0; r < 8; ++r)
      sS[(it * 16 + hl * 8 + r) * SPITCH + jt * 16 + ln] = acc[r];
  }
  __syncthreads();

  // ---- Phase 3: row softmax over key axis (one row per thread, float4) ----
  {
    float4* row4 = (float4*)(sS + tid * SPITCH);          // 16B-aligned rows
    float m = -3.0e38f;
#pragma unroll 4
    for (int j = 0; j < H_DIM / 4; ++j) {
      float4 v = row4[j];
      m = fmaxf(m, fmaxf(fmaxf(v.x, v.y), fmaxf(v.z, v.w)));
    }
    float s = 0.0f;
#pragma unroll 4
    for (int j = 0; j < H_DIM / 4; ++j) {
      float4 v = row4[j];
      v.x = __expf(v.x - m); v.y = __expf(v.y - m);
      v.z = __expf(v.z - m); v.w = __expf(v.w - m);
      s += (v.x + v.y) + (v.z + v.w);
      row4[j] = v;
    }
    const float inv = __frcp_rn(s);
#pragma unroll 4
    for (int j = 0; j < H_DIM / 4; ++j) {
      float4 v = row4[j];
      v.x *= inv; v.y *= inv; v.z *= inv; v.w *= inv;
      row4[j] = v;
    }
  }
  __syncthreads();

  // ---- Phase 4: Out^T[i][c] = sum_j att[i][j] * V[c][j]; fuse g*out + x ---
  for (int t = wave * 8; t < wave * 8 + 8; ++t) {
    const int it = t >> 2, ct = t & 3;
    v8f acc = {};
#pragma unroll
    for (int kk = 0; kk < H_DIM; kk += 32) {
      v16bf a = load_a_f32_rm(sS, it * 16, kk, SPITCH, lane);
      v16bf b = load_b_ct(sV, kk, ct * 16, H_DIM, lane);  // B[j][c]=sV[c*H+j]
      acc = wmma_bf16(a, b, acc);
    }
#pragma unroll
    for (int r = 0; r < 8; ++r) {
      const int i = it * 16 + hl * 8 + r;
      const int c = ct * 16 + ln;
      const size_t gidx = (((size_t)n * C_DIM + c) * H_DIM + i) * W_DIM + w;
      out[gidx] = g * acc[r] + x[gidx];
    }
  }
}

// ---------------------------------------------------------------------------
extern "C" void kernel_launch(void* const* d_in, const int* in_sizes, int n_in,
                              void* d_out, int out_size, void* d_ws, size_t ws_size,
                              hipStream_t stream) {
  const float* x    = (const float*)d_in[0];
  const float* wq   = (const float*)d_in[1];
  const float* wk   = (const float*)d_in[2];
  const float* wv   = (const float*)d_in[3];
  const float* gama = (const float*)d_in[4];
  float* out = (float*)d_out;

  // 98 KB dynamic LDS (scores aliased over staging): 3 WGs per 320 KB WGP.
  (void)hipFuncSetAttribute((const void*)column_attention_kernel,
                            hipFuncAttributeMaxDynamicSharedMemorySize,
                            (int)SMEM_BYTES);

  dim3 grid(N_BATCH * W_DIM);   // one workgroup per (n, w) column: 4096 WGs
  dim3 block(NTHREADS);         // 4 wave32
  column_attention_kernel<<<grid, block, SMEM_BYTES, stream>>>(x, wq, wk, wv,
                                                               gama, out);
}
